// SplitInjectionAttentionProcessor_46540265619593
// MI455X (gfx1250) — compile-verified, matrix operand order checked by reading, not run on hardware
//
#include <hip/hip_runtime.h>
#include <hip/hip_bf16.h>

// ---------------- problem constants ----------------
#define DM    1280
#define DC    768
#define S_LEN 4096
#define B_SZ  16
#define NH    8
#define DH    160      // head dim
#define NKV   48       // 3 * 16 kv tokens (anat | dis | delta)
#define NKVP  64       // padded kv for WMMA K=32 steps
#define TILE  64       // rows of S per block
#define NW    4        // waves per block (wave32)
#define TPB   (NW * 32)

// LDS row strides in f16 units (padded; all byte strides multiples of 16)
#define ZTS 1288       // z tile   [64][1288]
#define QHS 168        // q_head   [64][168]
#define PTS 72         // probs    [64][72]
#define BPS 40         // B panel  [160][40]  (32 data + 4-dword TDM pad)
#define PANEL_F16 (160 * BPS)
#define SMEM_F16 (TILE*ZTS + TILE*QHS + TILE*PTS + 2*PANEL_F16)

typedef __attribute__((ext_vector_type(16))) _Float16 v16h;
typedef __attribute__((ext_vector_type(8)))  _Float16 v8h;
typedef __attribute__((ext_vector_type(8)))  float    v8f;
typedef __attribute__((ext_vector_type(4)))  unsigned int u32x4;
typedef __attribute__((ext_vector_type(8)))  int i32x8;
typedef __attribute__((ext_vector_type(4)))  int i32x4;

__device__ __forceinline__ v16h cat8(v8h lo, v8h hi) {
  return __builtin_shufflevector(lo, hi, 0,1,2,3,4,5,6,7,8,9,10,11,12,13,14,15);
}

// Load a 16x32 f16 A/B fragment. For A: lane holds row (lane&15), K runs
// {half*8..+7} and {16+half*8..+7}. For B (stored N-major / transposed):
// identical addressing with row = n.
__device__ __forceinline__ v16h load_frag(const _Float16* base, int rs) {
  const int lane = threadIdx.x & 31;
  const int r  = lane & 15;
  const int hf = lane >> 4;
  const _Float16* p = base + r * rs + hf * 8;
  v8h lo = *(const v8h*)(p);
  v8h hi = *(const v8h*)(p + 16);
  return cat8(lo, hi);
}

__device__ __forceinline__ v8f wmma16(v16h a, v16h b, v8f c) {
  // D = A(16x32 f16) * B(32x16 f16) + C(16x16 f32)
  return __builtin_amdgcn_wmma_f32_16x16x32_f16(false, a, false, b, (short)0, c,
                                                false, false);
}

// ---- Tensor Data Mover: DMA a [160][32] f16 weight panel (row stride DM)
// from global into LDS, padding each 64B row with 16B so the LDS row stride
// becomes BPS(=40 f16) for bank-conflict-free fragment reads.
__device__ __forceinline__ void tdm_load_panel(const _Float16* src,
                                               _Float16* lds_dst) {
  const unsigned long long ga = (unsigned long long)(const void*)src;
  const unsigned lds_off = (unsigned)(unsigned long long)(void*)lds_dst;

  u32x4 g0;
  g0[0] = 1u;                                          // count=1, user mode
  g0[1] = lds_off;                                     // lds_addr (bytes)
  g0[2] = (unsigned)(ga & 0xFFFFFFFFu);                // global_addr[31:0]
  g0[3] = (unsigned)((ga >> 32) & 0x01FFFFFFu)         // global_addr[56:32]
          | (2u << 30);                                // type=2 (image)

  i32x8 g1;
  g1[0] = (1 << 16)        // data_size = 2 bytes
        | (1 << 20)        // pad_enable
        | (3 << 22)        // pad_interval: 16 dwords (one 64B tile row)
        | (3 << 25);       // pad_amount:   4 dwords (16B)
  g1[1] = (int)((DM & 0xFFFF) << 16);    // tensor_dim0[15:0]   (=1280)
  g1[2] = (int)((DM & 0xFFFF) << 16);    // dim0 hi=0 | tensor_dim1[15:0]
  g1[3] = (32 << 16);                    // dim1 hi=0 | tile_dim0 = 32
  g1[4] = 160;                           // tile_dim1 = 160, tile_dim2 = 0
  g1[5] = DM;                            // tensor_dim0_stride[31:0] = 1280
  g1[6] = 0;
  g1[7] = 0;

  i32x4 gz4 = {};
  i32x8 gz8 = {};
  __builtin_amdgcn_tensor_load_to_lds(g0, g1, gz4, gz4, gz8, 0);
}

// ---------------- prep kernels ----------------

// Transpose + fp32->f16 of w_q and w_out:  T[n][k] = W[k][n]
__global__ void prep_w_kernel(const float* __restrict__ wq,
                              const float* __restrict__ wout,
                              _Float16* __restrict__ wqT,
                              _Float16* __restrict__ woutT) {
  const int idx = blockIdx.x * blockDim.x + threadIdx.x;
  if (idx >= DM * DM) return;
  const int n = idx / DM;
  const int k = idx % DM;
  wqT[idx]   = (_Float16)wq[(size_t)k * DM + n];
  woutT[idx] = (_Float16)wout[(size_t)k * DM + n];
}

// K/V projections for the 48 concatenated kv tokens [anat | dis | delta].
// Kall: [B][48][1280] token-major (== B^T layout for the score GEMM).
// VallT: [B][1280][64] dim-major (== B^T layout for P@V), tokens 48..63 zero.
__global__ void kv_proj_kernel(const float* __restrict__ enc,
                               const float* __restrict__ wk,
                               const float* __restrict__ wv,
                               const float* __restrict__ wkd,
                               const float* __restrict__ wvd,
                               _Float16* __restrict__ Kall,
                               _Float16* __restrict__ VallT) {
  const int idx = blockIdx.x * blockDim.x + threadIdx.x;
  if (idx >= B_SZ * NKV * DM) return;
  const int d = idx % DM;
  const int t = (idx / DM) % NKV;
  const int b = idx / (DM * NKV);

  int srcTok;
  const float *WK, *WV;
  if (t < 16)      { srcTok = 16 + t; WK = wk;  WV = wv;  }   // anat
  else if (t < 32) { srcTok = t - 16; WK = wkd; WV = wvd; }   // dis
  else             { srcTok = t;      WK = wkd; WV = wvd; }   // delta

  const float* e = enc + ((size_t)b * NKV + srcTok) * DC;
  float ak = 0.f, av = 0.f;
  for (int c = 0; c < DC; ++c) {
    const float ev = e[c];
    ak += ev * WK[(size_t)c * DM + d];
    av += ev * WV[(size_t)c * DM + d];
  }
  Kall[idx] = (_Float16)ak;
  VallT[((size_t)b * DM + d) * NKVP + t] = (_Float16)av;
  if (t < 16) VallT[((size_t)b * DM + d) * NKVP + 48 + t] = (_Float16)0.0f;
}

// ---------------- fused attention kernel ----------------
__global__ void __launch_bounds__(TPB)
attn_fused_kernel(const float*    __restrict__ hs,
                  const _Float16* __restrict__ wqT,
                  const _Float16* __restrict__ woutT,
                  const _Float16* __restrict__ Kall,
                  const _Float16* __restrict__ VallT,
                  const float*    __restrict__ b_out,
                  float*          __restrict__ out) {
  extern __shared__ __align__(16) char smem_raw[];
  _Float16* zt = (_Float16*)smem_raw;        // [64][ZTS]  f16 z tile
  _Float16* qh = zt + TILE * ZTS;            // [64][QHS]  q for current head
  _Float16* pt = qh + TILE * QHS;            // [64][PTS]  probs (padded K=64)
  _Float16* bp = pt + TILE * PTS;            // [2][160][BPS] double-buffered panel

  const int sBase = blockIdx.x * TILE;
  const int bIdx  = blockIdx.y;
  const int lane  = threadIdx.x & 31;
  const int wave  = threadIdx.x >> 5;
  const int l16   = lane & 15;
  const int hf    = lane >> 4;

  const float scale = 0.07905694150420949f;  // 1/sqrt(160)
  const int KSTEPS = DM / 32;                // 40

  // each lane's A-matrix row of hidden_states (fp32)
  const float* hsRow =
      hs + ((size_t)bIdx * S_LEN + sBase + wave * 16 + l16) * DM;

  for (int h = 0; h < NH; ++h) {
    // ---------- Q projection: q_h(64x160) = hs(64x1280) @ wq[:,h] ----------
    const _Float16* wsrc = wqT + (size_t)(h * DH) * DM;
    if (wave == 0) tdm_load_panel(wsrc, bp);           // prime panel kc=0
    v8f accq[10] = {};
    for (int kc = 0; kc < KSTEPS; ++kc) {
      if (wave == 0) {
        if (kc + 1 < KSTEPS) {
          tdm_load_panel(wsrc + (kc + 1) * 32, bp + ((kc + 1) & 1) * PANEL_F16);
          __builtin_amdgcn_s_wait_tensorcnt(1);   // panel kc done, kc+1 in flight
        } else {
          __builtin_amdgcn_s_wait_tensorcnt(0);   // drain
        }
      }
      __syncthreads();                       // panel kc visible to all waves
      const _Float16* bpc = bp + (kc & 1) * PANEL_F16;
      v16h a;
      {
        const float* p0 = hsRow + kc * 32 + hf * 8;
        #pragma unroll
        for (int i = 0; i < 8; ++i) {
          a[i]     = (_Float16)p0[i];
          a[8 + i] = (_Float16)p0[16 + i];
        }
      }
      #pragma unroll
      for (int nt = 0; nt < 10; ++nt) {
        v16h bf = load_frag(bpc + nt * 16 * BPS, BPS);
        accq[nt] = wmma16(a, bf, accq[nt]);
      }
      __syncthreads();                       // done reading panel kc
    }
    // spill q_h to LDS as f16 (C layout: row = r + 8*hf, col = l16)
    #pragma unroll
    for (int nt = 0; nt < 10; ++nt) {
      #pragma unroll
      for (int r = 0; r < 8; ++r) {
        qh[(wave * 16 + r + 8 * hf) * QHS + nt * 16 + l16] =
            (_Float16)accq[nt][r];
      }
    }
    __syncthreads();

    // ---------- scores(64x48) = q_h @ Kall_h^T ----------
    v8f accs[3] = {};
    #pragma unroll
    for (int kc = 0; kc < DH / 32; ++kc) {
      v16h a = load_frag(qh + (wave * 16) * QHS + kc * 32, QHS);
      #pragma unroll
      for (int t = 0; t < 3; ++t) {
        const _Float16* kp = Kall +
            ((size_t)(bIdx * NKV + t * 16 + l16)) * DM + h * DH + kc * 32 + hf * 8;
        v16h bf = cat8(*(const v8h*)kp, *(const v8h*)(kp + 16));
        accs[t] = wmma16(a, bf, accs[t]);
      }
    }

    // ---------- softmax per 16-key branch, gates folded in ----------
    #pragma unroll
    for (int t = 0; t < 3; ++t) {
      const float gate = (t == 2) ? 1.0f : 0.5f;
      #pragma unroll
      for (int r = 0; r < 8; ++r) {
        const float v = accs[t][r] * scale;
        float m = v;
        #pragma unroll
        for (int off = 1; off < 16; off <<= 1)
          m = fmaxf(m, __shfl_xor(m, off, 16));
        const float e = __expf(v - m);
        float s = e;
        #pragma unroll
        for (int off = 1; off < 16; off <<= 1)
          s += __shfl_xor(s, off, 16);
        pt[(wave * 16 + r + 8 * hf) * PTS + t * 16 + l16] =
            (_Float16)(e / s * gate);
      }
    }
    #pragma unroll
    for (int r = 0; r < 8; ++r)   // zero the padded tokens 48..63
      pt[(wave * 16 + r + 8 * hf) * PTS + 48 + l16] = (_Float16)0.0f;
    __syncthreads();

    // ---------- z_h(64x160) = P(64x64) @ Vall_h(64x160) ----------
    v8f accz[10] = {};
    #pragma unroll
    for (int kc = 0; kc < 2; ++kc) {
      v16h a = load_frag(pt + (wave * 16) * PTS + kc * 32, PTS);
      #pragma unroll
      for (int nt = 0; nt < 10; ++nt) {
        const _Float16* vp = VallT +
            ((size_t)bIdx * DM + h * DH + nt * 16 + l16) * NKVP + kc * 32 + hf * 8;
        v16h bf = cat8(*(const v8h*)vp, *(const v8h*)(vp + 16));
        accz[nt] = wmma16(a, bf, accz[nt]);
      }
    }
    #pragma unroll
    for (int nt = 0; nt < 10; ++nt) {
      #pragma unroll
      for (int r = 0; r < 8; ++r) {
        zt[(wave * 16 + r + 8 * hf) * ZTS + h * DH + nt * 16 + l16] =
            (_Float16)accz[nt][r];
      }
    }
    __syncthreads();
  }

  // ---------- out = z(64x1280) @ w_out + b_out + residual ----------
  for (int ng = 0; ng < NH; ++ng) {        // 8 column groups of 160
    const _Float16* wsrc = woutT + (size_t)(ng * DH) * DM;
    if (wave == 0) tdm_load_panel(wsrc, bp);
    v8f acco[10] = {};
    for (int kc = 0; kc < KSTEPS; ++kc) {
      if (wave == 0) {
        if (kc + 1 < KSTEPS) {
          tdm_load_panel(wsrc + (kc + 1) * 32, bp + ((kc + 1) & 1) * PANEL_F16);
          __builtin_amdgcn_s_wait_tensorcnt(1);
        } else {
          __builtin_amdgcn_s_wait_tensorcnt(0);
        }
      }
      __syncthreads();
      const _Float16* bpc = bp + (kc & 1) * PANEL_F16;
      v16h a = load_frag(zt + (wave * 16) * ZTS + kc * 32, ZTS);
      #pragma unroll
      for (int nt = 0; nt < 10; ++nt) {
        v16h bf = load_frag(bpc + nt * 16 * BPS, BPS);
        acco[nt] = wmma16(a, bf, acco[nt]);
      }
      __syncthreads();
    }
    #pragma unroll
    for (int nt = 0; nt < 10; ++nt) {
      #pragma unroll
      for (int r = 0; r < 8; ++r) {
        const int row = wave * 16 + r + 8 * hf;
        const int col = ng * DH + nt * 16 + l16;
        const size_t g = ((size_t)bIdx * S_LEN + sBase + row) * DM + col;
        out[g] = acco[nt][r] + b_out[col] + hs[g];
      }
    }
  }
}

// ---------------- launcher ----------------
extern "C" void kernel_launch(void* const* d_in, const int* in_sizes, int n_in,
                              void* d_out, int out_size, void* d_ws, size_t ws_size,
                              hipStream_t stream) {
  (void)in_sizes; (void)n_in; (void)out_size; (void)ws_size;
  const float* hs   = (const float*)d_in[0];
  const float* enc  = (const float*)d_in[1];
  const float* w_q  = (const float*)d_in[2];
  const float* w_k  = (const float*)d_in[3];
  const float* w_v  = (const float*)d_in[4];
  const float* w_kd = (const float*)d_in[5];
  const float* w_vd = (const float*)d_in[6];
  const float* w_o  = (const float*)d_in[7];
  const float* b_o  = (const float*)d_in[8];
  float* out = (float*)d_out;

  _Float16* ws    = (_Float16*)d_ws;
  _Float16* wqT   = ws;
  _Float16* woutT = wqT + (size_t)DM * DM;
  _Float16* Kall  = woutT + (size_t)DM * DM;
  _Float16* VallT = Kall + (size_t)B_SZ * NKV * DM;

  prep_w_kernel<<<(DM * DM + 255) / 256, 256, 0, stream>>>(w_q, w_o, wqT, woutT);
  kv_proj_kernel<<<(B_SZ * NKV * DM + 255) / 256, 256, 0, stream>>>(
      enc, w_k, w_v, w_kd, w_vd, Kall, VallT);

  dim3 grid(S_LEN / TILE, B_SZ);
  const size_t lds_bytes = (size_t)SMEM_F16 * sizeof(_Float16);
  attn_fused_kernel<<<grid, TPB, lds_bytes, stream>>>(hs, wqT, woutT, Kall,
                                                      VallT, b_o, out);
}
